// Rel_Multihead_Att_532575945175
// MI455X (gfx1250) — matrix operand order repaired and checked by reading, not compiled, not run on hardware
//
#include <hip/hip_runtime.h>

// Transformer-XL relative multi-head attention for MI455X (gfx1250).
// All GEMMs via v_wmma_f32_16x16x32_bf16 (wave32). Weights pre-transposed so
// both A and B WMMA fragments are contiguous 16B global loads. GEMM k-loop is
// register double-buffered so global_load_b128 clauses overlap WMMA issue.
// Workspace use: ~150 MB (x/pos/W bf16 copies, q/k/v, BD_raw 64MB, preln).

#define B_ 4
#define Q_ 1024
#define H_ 1024
#define N_ 8
#define D_ 128
#define SCALE_ 0.08838834764831845f   // 1/sqrt(128)

typedef __attribute__((ext_vector_type(16))) __bf16        v16bf;
typedef __attribute__((ext_vector_type(8)))  float         v8f;
typedef __attribute__((ext_vector_type(4)))  unsigned int  v4u;

union FragBF {
    v16bf v;
    unsigned short u[16];
    v4u q[2];
};

__device__ inline unsigned short f2bf(float f) {
    unsigned u = __float_as_uint(f);
    u = u + 0x7FFFu + ((u >> 16) & 1u);          // round-to-nearest-even
    return (unsigned short)(u >> 16);
}
__device__ inline float bf2f(unsigned short h) {
    return __uint_as_float(((unsigned)h) << 16);
}
__device__ inline v8f wmma_bf16(v16bf a, v16bf b, v8f c) {
    // (neg_a, A, neg_b, B, c_mod, C, reuse_a, reuse_b)
    return __builtin_amdgcn_wmma_f32_16x16x32_bf16(false, a, false, b, (short)0, c, false, false);
}

// ---------------------------------------------------------------- converts
__global__ __launch_bounds__(256)
void k_cvt(const float* __restrict__ in, unsigned short* __restrict__ out, int n) {
    int i = blockIdx.x * 256 + threadIdx.x;
    if (i < n) out[i] = f2bf(in[i]);
}

// in: [R,C] row-major f32  ->  out: [C,R] row-major bf16 (transpose)
__global__ __launch_bounds__(256)
void k_cvtT(const float* __restrict__ in, unsigned short* __restrict__ out, int R, int C) {
    int i = blockIdx.x * 256 + threadIdx.x;
    if (i >= R * C) return;
    int r = i / C, c = i % C;
    out[(long)c * R + r] = f2bf(in[i]);
}

// ---------------------------------------------------------------- generic WMMA GEMM
// C[M,N] = A[M,K] @ Bt[N,K]^T.  Grid: (N/256, M/64, batches). 256 thr = 8 waves
// arranged 2x4; wave tile = 32 rows x 64 cols (2x4 accumulators).
// K-loop double-buffered in registers (ping-pong fragment sets, step 64).
// Batch offsets: (z/8)*s0 + (z%8)*s1.
// mode 0: out0 = bf16(acc)
// mode 1: out0 = bf16(acc+bias0[col]); out1 = bf16(acc+bias1[col])
// mode 2: outf = acc + resid[row*ldo+col]          (f32)
// mode 3: kv split -> out0 = K [BQ,H] bf16; out1 = V transposed [B,N,D,Q] bf16
__global__ __launch_bounds__(256)
void k_gemm(const unsigned short* __restrict__ A,
            const unsigned short* __restrict__ Bt,
            int K, int lda, int ldbt, int ldo,
            long a0, long a1, long b0, long b1, long c0, long c1,
            int mode,
            unsigned short* __restrict__ out0,
            unsigned short* __restrict__ out1,
            const float* __restrict__ bias0,
            const float* __restrict__ bias1,
            const float* __restrict__ resid,
            float* __restrict__ outf) {
    int z = blockIdx.z;
    A  += (long)(z / N_) * a0 + (long)(z % N_) * a1;
    Bt += (long)(z / N_) * b0 + (long)(z % N_) * b1;
    long cOff = (long)(z / N_) * c0 + (long)(z % N_) * c1;

    int lane = threadIdx.x & 31;
    int w = threadIdx.x >> 5;
    int wr = w & 1, wc = w >> 1;          // 2 x 4 wave grid
    int mb = blockIdx.y * 64 + wr * 32;   // wave: 32 rows
    int nb = blockIdx.x * 256 + wc * 64;  // wave: 64 cols
    int l16 = lane & 15;
    int hi = lane >> 4;

    v8f acc[2][4];
    for (int i = 0; i < 2; i++)
        for (int j = 0; j < 4; j++)
            for (int e = 0; e < 8; e++) acc[i][j][e] = 0.f;

    const unsigned short* aRow[2];
    aRow[0] = A + (long)(mb + l16) * lda + hi * 8;
    aRow[1] = A + (long)(mb + 16 + l16) * lda + hi * 8;
    const unsigned short* bRow[4];
#pragma unroll
    for (int j = 0; j < 4; j++)
        bRow[j] = Bt + (long)(nb + j * 16 + l16) * ldbt + hi * 8;

    FragBF fa[2][2];      // [buf][rowtile]
    FragBF fb[2][4];      // [buf][coltile]

    auto loadAB = [&](int kk, int buf) {
#pragma unroll
        for (int i = 0; i < 2; i++) {
            fa[buf][i].q[0] = *(const v4u*)(aRow[i] + kk);        // K +0..7 / +8..15
            fa[buf][i].q[1] = *(const v4u*)(aRow[i] + kk + 16);   // K +16..23 / +24..31
        }
#pragma unroll
        for (int j = 0; j < 4; j++) {
            fb[buf][j].q[0] = *(const v4u*)(bRow[j] + kk);        // K +0..7 / +8..15
            fb[buf][j].q[1] = *(const v4u*)(bRow[j] + kk + 16);
        }
    };
    auto mmaAll = [&](int buf) {
#pragma unroll
        for (int i = 0; i < 2; i++)
#pragma unroll
            for (int j = 0; j < 4; j++)
                acc[i][j] = wmma_bf16(fa[buf][i].v, fb[buf][j].v, acc[i][j]);
    };

    // software pipeline, step 64 (K is always a multiple of 64 here)
    loadAB(0, 0);
    int kk = 0;
    for (; kk + 64 < K; kk += 64) {
        loadAB(kk + 32, 1);
        mmaAll(0);
        loadAB(kk + 64, 0);
        mmaAll(1);
    }
    loadAB(kk + 32, 1);
    mmaAll(0);
    mmaAll(1);

#pragma unroll
    for (int i = 0; i < 2; i++) {
#pragma unroll
        for (int j = 0; j < 4; j++) {
#pragma unroll
            for (int r = 0; r < 8; r++) {
                int row = mb + i * 16 + r + hi * 8;
                int col = nb + j * 16 + l16;
                float v = acc[i][j][r];
                if (mode == 0) {
                    out0[cOff + (long)row * ldo + col] = f2bf(v);
                } else if (mode == 1) {
                    out0[(long)row * ldo + col] = f2bf(v + bias0[col]);
                    out1[(long)row * ldo + col] = f2bf(v + bias1[col]);
                } else if (mode == 2) {
                    outf[(long)row * ldo + col] = v + resid[(long)row * ldo + col];
                } else {                           // mode 3: kv split
                    if (col < H_) {
                        out0[(long)row * H_ + col] = f2bf(v);        // K matrix
                    } else {
                        int c2 = col - H_;
                        int n = c2 / D_, d = c2 % D_;
                        int b = row / Q_, qq = row % Q_;
                        out1[(((long)b * N_ + n) * D_ + d) * Q_ + qq] = f2bf(v);  // V^T
                    }
                }
            }
        }
    }
}

// ---------------------------------------------------------------- flash attention
// Grid: (Q/64, B*N), 128 thr = 4 waves, wave owns 16 query rows.
// score[q,k] = ((q+rw)·k + BD_raw[q, Q-1-q+k]) * SCALE for k<=q, else -inf.
__global__ __launch_bounds__(128)
void k_attn(const unsigned short* __restrict__ qw,
            const unsigned short* __restrict__ kmat,
            const unsigned short* __restrict__ vT,
            const unsigned short* __restrict__ bd,
            unsigned short* __restrict__ att) {
    __shared__ unsigned short ldsP[4][16][32];

    int lane = threadIdx.x & 31;
    int w = threadIdx.x >> 5;
    int l16 = lane & 15;
    int hi = lane >> 4;
    int qb = blockIdx.x * 64 + w * 16;
    int b = blockIdx.y >> 3;
    int n = blockIdx.y & 7;

    // preload query fragments (q + rw_bias), 4 chunks of K=32 over D=128
    FragBF fq[4];
    {
        const unsigned short* base = qw + ((long)b * Q_ + qb + l16) * H_ + n * D_;
#pragma unroll
        for (int c = 0; c < 4; c++) {
            const unsigned short* p = base + c * 32 + hi * 8;
            fq[c].q[0] = *(const v4u*)p;
            fq[c].q[1] = *(const v4u*)(p + 16);
        }
    }

    v8f O[8];
    for (int t = 0; t < 8; t++)
        for (int e = 0; e < 8; e++) O[t][e] = 0.f;
    float mrow[8], lrow[8];
    for (int r = 0; r < 8; r++) { mrow[r] = -1e30f; lrow[r] = 0.f; }

    const long bdbase = ((long)b * N_ + n) * Q_ * Q_;
    const unsigned short* krow = kmat + ((long)b * Q_ + l16) * H_ + n * D_;
    const unsigned short* vrow = vT + (((long)b * N_ + n) * D_ + l16) * Q_;

    for (int kt = 0; kt < qb + 16; kt += 32) {
        // ---- scores: two 16-wide key tiles, AC via WMMA (batched frag loads)
        v8f S[2];
        for (int t = 0; t < 2; t++)
            for (int e = 0; e < 8; e++) S[t][e] = 0.f;
#pragma unroll
        for (int t = 0; t < 2; t++) {
            FragBF fk[4];
#pragma unroll
            for (int c = 0; c < 4; c++) {
                const unsigned short* p = krow + (long)(kt + t * 16) * H_ + c * 32 + hi * 8;
                fk[c].q[0] = *(const v4u*)p;
                fk[c].q[1] = *(const v4u*)(p + 16);
            }
#pragma unroll
            for (int c = 0; c < 4; c++)
                S[t] = wmma_bf16(fq[c].v, fk[c].v, S[t]);
        }
        // ---- add rel-position term (gather), scale, causal mask
        float tmax[8];
        for (int r = 0; r < 8; r++) tmax[r] = -1e30f;
#pragma unroll
        for (int t = 0; t < 2; t++) {
#pragma unroll
            for (int r = 0; r < 8; r++) {
                int q = qb + r + hi * 8;
                int k = kt + t * 16 + l16;
                float s;
                if (k <= q) {
                    float bdv = bf2f(bd[bdbase + (long)q * Q_ + (Q_ - 1 - q + k)]);
                    s = (S[t][r] + bdv) * SCALE_;
                } else {
                    s = -1e30f;
                }
                S[t][r] = s;
                tmax[r] = fmaxf(tmax[r], s);
            }
        }
        // ---- row max across 16 lanes (halves stay separate w/ masks < 16)
        for (int r = 0; r < 8; r++) {
            float v = tmax[r];
            v = fmaxf(v, __shfl_xor(v, 1, 32));
            v = fmaxf(v, __shfl_xor(v, 2, 32));
            v = fmaxf(v, __shfl_xor(v, 4, 32));
            v = fmaxf(v, __shfl_xor(v, 8, 32));
            tmax[r] = v;
        }
        // ---- online softmax update
        float alpha[8], rsum[8];
        for (int r = 0; r < 8; r++) {
            float mnew = fmaxf(mrow[r], tmax[r]);
            alpha[r] = __expf(mrow[r] - mnew);
            mrow[r] = mnew;
            rsum[r] = 0.f;
        }
#pragma unroll
        for (int t = 0; t < 2; t++) {
#pragma unroll
            for (int r = 0; r < 8; r++) {
                float p = __expf(S[t][r] - mrow[r]);
                S[t][r] = p;
                rsum[r] += p;
            }
        }
        for (int r = 0; r < 8; r++) {
            float v = rsum[r];
            v += __shfl_xor(v, 1, 32);
            v += __shfl_xor(v, 2, 32);
            v += __shfl_xor(v, 4, 32);
            v += __shfl_xor(v, 8, 32);
            lrow[r] = lrow[r] * alpha[r] + v;
        }
#pragma unroll
        for (int t = 0; t < 8; t++)
#pragma unroll
            for (int r = 0; r < 8; r++) O[t][r] *= alpha[r];

        // ---- P (C-layout) -> LDS -> A-fragment layout (wave-private staging)
#pragma unroll
        for (int t = 0; t < 2; t++)
#pragma unroll
            for (int r = 0; r < 8; r++)
                ldsP[w][r + hi * 8][t * 16 + l16] = f2bf(S[t][r]);
        FragBF fp;
        {
            const unsigned short* lp = &ldsP[w][l16][hi * 8];
            fp.q[0] = *(const v4u*)lp;
            fp.q[1] = *(const v4u*)(lp + 16);
        }
        // ---- O += P @ V  (V transposed: contiguous K per lane; 2 batches of 4)
#pragma unroll
        for (int g = 0; g < 2; g++) {
            FragBF fv[4];
#pragma unroll
            for (int c = 0; c < 4; c++) {
                const unsigned short* p = vrow + (long)(g * 4 + c) * 16 * Q_ + kt + hi * 8;
                fv[c].q[0] = *(const v4u*)p;
                fv[c].q[1] = *(const v4u*)(p + 16);
            }
#pragma unroll
            for (int c = 0; c < 4; c++)
                O[g * 4 + c] = wmma_bf16(fp.v, fv[c].v, O[g * 4 + c]);
        }
    }

    // ---- normalize and store attended [BQ, N*D] bf16
#pragma unroll
    for (int dt = 0; dt < 8; dt++) {
#pragma unroll
        for (int r = 0; r < 8; r++) {
            int q = qb + r + hi * 8;
            int col = n * D_ + dt * 16 + l16;
            att[((long)b * Q_ + q) * H_ + col] = f2bf(O[dt][r] / lrow[r]);
        }
    }
}

// ---------------------------------------------------------------- layer norm
__global__ __launch_bounds__(256)
void k_ln(const float* __restrict__ xin, const float* __restrict__ g,
          const float* __restrict__ beta, float* __restrict__ out) {
    __shared__ float red[8];
    int row = blockIdx.x;
    const float* p = xin + (long)row * H_;
    int tid = threadIdx.x;
    int w = tid >> 5, lane = tid & 31;

    float s = 0.f;
    for (int i = tid; i < H_; i += 256) s += p[i];
    for (int m = 1; m < 32; m <<= 1) s += __shfl_xor(s, m, 32);
    if (lane == 0) red[w] = s;
    __syncthreads();
    float tot = 0.f;
    for (int i = 0; i < 8; i++) tot += red[i];
    float mu = tot * (1.f / H_);
    __syncthreads();

    float v = 0.f;
    for (int i = tid; i < H_; i += 256) { float d = p[i] - mu; v += d * d; }
    for (int m = 1; m < 32; m <<= 1) v += __shfl_xor(v, m, 32);
    if (lane == 0) red[w] = v;
    __syncthreads();
    float vt = 0.f;
    for (int i = 0; i < 8; i++) vt += red[i];
    float inv = rsqrtf(vt * (1.f / H_) + 1e-5f);

    for (int i = tid; i < H_; i += 256)
        out[(long)row * H_ + i] = (p[i] - mu) * inv * g[i] + beta[i];
}

// ---------------------------------------------------------------- launch
extern "C" void kernel_launch(void* const* d_in, const int* in_sizes, int n_in,
                              void* d_out, int out_size, void* d_ws, size_t ws_size,
                              hipStream_t stream) {
    const float* x       = (const float*)d_in[0];
    /* d_in[1] = mask (bool) — causal structure handled analytically */
    const float* pos_emb = (const float*)d_in[2];
    const float* rr_bias = (const float*)d_in[3];
    const float* rw_bias = (const float*)d_in[4];
    const float* W_kv    = (const float*)d_in[5];
    const float* W_q     = (const float*)d_in[6];
    const float* W_r     = (const float*)d_in[7];
    const float* W_o     = (const float*)d_in[8];
    const float* ln_g    = (const float*)d_in[9];
    const float* ln_b    = (const float*)d_in[10];
    float* out = (float*)d_out;

    char* ws = (char*)d_ws;
    auto take = [&](size_t bytes) -> char* {
        char* p = ws;
        ws += (bytes + 255) & ~(size_t)255;
        return p;
    };
    const size_t BQ = (size_t)B_ * Q_;
    unsigned short* x_bf   = (unsigned short*)take(BQ * H_ * 2);
    unsigned short* pos_bf = (unsigned short*)take((size_t)Q_ * H_ * 2);
    unsigned short* WkvT   = (unsigned short*)take((size_t)2 * H_ * H_ * 2);
    unsigned short* WqT    = (unsigned short*)take((size_t)H_ * H_ * 2);
    unsigned short* WrT    = (unsigned short*)take((size_t)H_ * H_ * 2);
    unsigned short* WoT    = (unsigned short*)take((size_t)H_ * H_ * 2);
    unsigned short* k_bf   = (unsigned short*)take(BQ * H_ * 2);
    unsigned short* vT     = (unsigned short*)take(BQ * H_ * 2);
    unsigned short* qw     = (unsigned short*)take(BQ * H_ * 2);
    unsigned short* qr     = (unsigned short*)take(BQ * H_ * 2);
    unsigned short* r_bf   = (unsigned short*)take((size_t)Q_ * H_ * 2);
    unsigned short* bd     = (unsigned short*)take((size_t)B_ * N_ * Q_ * Q_ * 2);
    unsigned short* att    = (unsigned short*)take(BQ * H_ * 2);
    float*          preln  = (float*)take(BQ * H_ * 4);

    // 1) converts (weights transposed so GEMM B-operand is column-major)
    k_cvt <<<dim3((int)(BQ * H_ / 256)), 256, 0, stream>>>(x, x_bf, (int)(BQ * H_));
    k_cvt <<<dim3(Q_ * H_ / 256), 256, 0, stream>>>(pos_emb, pos_bf, Q_ * H_);
    k_cvtT<<<dim3(H_ * 2 * H_ / 256), 256, 0, stream>>>(W_kv, WkvT, H_, 2 * H_);
    k_cvtT<<<dim3(H_ * H_ / 256), 256, 0, stream>>>(W_q, WqT, H_, H_);
    k_cvtT<<<dim3(H_ * H_ / 256), 256, 0, stream>>>(W_r, WrT, H_, H_);
    k_cvtT<<<dim3(H_ * H_ / 256), 256, 0, stream>>>(W_o, WoT, H_, H_);

    // 2) kv = x @ W_kv  -> K [BQ,H] bf16 + V^T [B,N,D,Q] bf16   (mode 3)
    k_gemm<<<dim3(2 * H_ / 256, (int)(BQ / 64), 1), 256, 0, stream>>>(
        x_bf, WkvT, H_, H_, H_, 2 * H_, 0, 0, 0, 0, 0, 0,
        3, k_bf, vT, nullptr, nullptr, nullptr, nullptr);

    // 3) q = x @ W_q; qw = q+rw_bias, qr = q+rr_bias  (mode 1)
    k_gemm<<<dim3(H_ / 256, (int)(BQ / 64), 1), 256, 0, stream>>>(
        x_bf, WqT, H_, H_, H_, H_, 0, 0, 0, 0, 0, 0,
        1, qw, qr, rw_bias, rr_bias, nullptr, nullptr);

    // 4) r = pos_emb @ W_r  (mode 0)
    k_gemm<<<dim3(H_ / 256, Q_ / 64, 1), 256, 0, stream>>>(
        pos_bf, WrT, H_, H_, H_, H_, 0, 0, 0, 0, 0, 0,
        0, r_bf, nullptr, nullptr, nullptr, nullptr, nullptr);

    // 5) BD_raw[b,n,q,j] = qr[b,q,n,:] · r[j,n,:]   batched over z = b*N+n
    k_gemm<<<dim3(Q_ / 256, Q_ / 64, B_ * N_), 256, 0, stream>>>(
        qr, r_bf, D_, H_, H_, Q_,
        (long)Q_ * H_, (long)D_,            // A offsets: b -> full slab, n -> +n*D
        0, (long)D_,                        // Bt offsets: n -> +n*D
        (long)N_ * Q_ * Q_, (long)Q_ * Q_,  // C offsets
        0, bd, nullptr, nullptr, nullptr, nullptr, nullptr);

    // 6) flash attention  ->  attended [BQ, N*D] bf16
    k_attn<<<dim3(Q_ / 64, B_ * N_), 128, 0, stream>>>(qw, k_bf, vT, bd, att);

    // 7) out = attended @ W_o + x  (mode 2, f32)
    k_gemm<<<dim3(H_ / 256, (int)(BQ / 64), 1), 256, 0, stream>>>(
        att, WoT, H_, H_, H_, H_, 0, 0, 0, 0, 0, 0,
        2, nullptr, nullptr, nullptr, nullptr, x, preln);

    // 8) layer norm -> d_out
    k_ln<<<dim3((int)BQ), 256, 0, stream>>>(preln, ln_g, ln_b, out);
}
